// SinkhornTransformerModule_44401371906398
// MI455X (gfx1250) — compile-verified, matrix operand order checked by reading, not run on hardware
//
#include <hip/hip_runtime.h>
#include <math.h>

typedef __bf16 bf16;
typedef __bf16 v8bf  __attribute__((ext_vector_type(8)));
typedef __bf16 v16bf __attribute__((ext_vector_type(16)));
typedef float  v8f   __attribute__((ext_vector_type(8)));

// pointer types expected by __builtin_amdgcn_global_load_async_to_lds_b128:
//   (int4 addrspace(1)*, int4 addrspace(3)*, imm offset, imm cpol)
typedef int int4v __attribute__((vector_size(16)));
typedef __attribute__((address_space(1))) int4v gbl_i4;
typedef __attribute__((address_space(3))) int4v lds_i4;

// generic LDS pointer -> AS(3) pointer (low 32 bits of generic LDS address are the LDS offset)
static __device__ __forceinline__ lds_i4* lds_cast(void* p) {
  return (lds_i4*)(uint32_t)(uintptr_t)p;
}
static __device__ __forceinline__ gbl_i4* gbl_cast(const void* p) {
  return (gbl_i4*)(uintptr_t)p;
}

#define E_DIM   512
#define T_TOK   8192
#define S_TOK   8190
#define NHEAD   8
#define NBUK    128
#define NLAY    8
#define VOCAB   17

#define GEMM_OUT_F32   0
#define GEMM_GELU_BF16 1
#define GEMM_ADD_RES   2

// ---------------------------------------------------------------- embeddings
__global__ void sinkformer_embed_kernel(const int* __restrict__ value,
                                        const int* __restrict__ depth,
                                        const int* __restrict__ pos,
                                        const float* __restrict__ sos,
                                        const float* __restrict__ tok,
                                        const float* __restrict__ dep,
                                        const float* __restrict__ spat,
                                        float* __restrict__ x) {
  const int t = blockIdx.x;
  for (int e = threadIdx.x; e < E_DIM; e += blockDim.x) {
    float v;
    if (t == 0) {
      v = sos[e];
    } else {
      const int src = t - 1;
      if (src < S_TOK) {
        v = tok[value[src] * E_DIM + e] + dep[depth[src] * E_DIM + e];
#pragma unroll
        for (int a = 0; a < 3; ++a)
          v += spat[(a * 65 + pos[a * S_TOK + src]) * E_DIM + e];
      } else {
        v = 0.f;  // padded tokens hit the zeroed rows in the reference
      }
    }
    x[(size_t)t * E_DIM + e] = v;
  }
}

// ---------------------------------------------------------------- layernorm -> bf16
__global__ __launch_bounds__(256) void sinkformer_ln_kernel(const float* __restrict__ x,
                                                            const float* __restrict__ g,
                                                            const float* __restrict__ b,
                                                            bf16* __restrict__ out) {
  __shared__ float red[8];
  const int t = blockIdx.x;
  const float* row = x + (size_t)t * E_DIM;
  const int e0 = threadIdx.x, e1 = threadIdx.x + 256;
  float v0 = row[e0], v1 = row[e1];
  float s = v0 + v1;
#pragma unroll
  for (int m = 16; m; m >>= 1) s += __shfl_xor(s, m, 32);
  const int wid = threadIdx.x >> 5, lid = threadIdx.x & 31;
  if (lid == 0) red[wid] = s;
  __syncthreads();
  float tot = 0.f;
#pragma unroll
  for (int i = 0; i < 8; ++i) tot += red[i];
  const float mu = tot * (1.f / 512.f);
  const float d0 = v0 - mu, d1 = v1 - mu;
  float q = d0 * d0 + d1 * d1;
#pragma unroll
  for (int m = 16; m; m >>= 1) q += __shfl_xor(q, m, 32);
  __syncthreads();
  if (lid == 0) red[wid] = q;
  __syncthreads();
  float vtot = 0.f;
#pragma unroll
  for (int i = 0; i < 8; ++i) vtot += red[i];
  const float rs = rsqrtf(vtot * (1.f / 512.f) + 1e-5f);
  out[(size_t)t * E_DIM + e0] = (bf16)(d0 * rs * g[e0] + b[e0]);
  out[(size_t)t * E_DIM + e1] = (bf16)(d1 * rs * g[e1] + b[e1]);
}

// ---------------------------------------------------------------- fp32 W(KxN) -> bf16 Wt(NxK)
__global__ void sinkformer_convt_kernel(const float* __restrict__ W, bf16* __restrict__ Wt,
                                        int K, int N) {
  const int idx = blockIdx.x * blockDim.x + threadIdx.x;
  if (idx >= K * N) return;
  const int k = idx / N, n = idx - k * N;
  Wt[(size_t)n * K + k] = (bf16)W[idx];
}

// ---------------------------------------------------------------- WMMA GEMM  C(MxN)=A(MxK)*Bt(NxK)^T
// block tile 128x64, 8 waves as 4x2 grid of 32x32 wave tiles.
// K staged 64 deep, DOUBLE-BUFFERED via async global->LDS DMA (ASYNCcnt).
__global__ __launch_bounds__(256) void sinkformer_gemm_kernel(
    const bf16* __restrict__ A, const bf16* __restrict__ Bt,
    const float* __restrict__ bias, const float* residual,
    float* outF, bf16* outB, int M, int N, int K, int mode) {
  __shared__ bf16 As[2][128 * 64];  // 2 x 16KB
  __shared__ bf16 Bs[2][64 * 64];   // 2 x 8KB
  const int tid = threadIdx.x;
  const int lane = tid & 31;
  const int wid = tid >> 5;
  const int waveM = wid >> 1;  // 0..3
  const int waveN = wid & 1;   // 0..1
  const int m0 = blockIdx.y * 128;
  const int n0 = blockIdx.x * 64;

  v8f acc[2][2];
#pragma unroll
  for (int i = 0; i < 2; ++i)
#pragma unroll
    for (int j = 0; j < 2; ++j)
#pragma unroll
      for (int e = 0; e < 8; ++e) acc[i][j][e] = 0.f;

  const int kb = (lane < 16) ? 0 : 8;
  const int mL = lane & 15;
  const int kTiles = K >> 6;

  // per-thread async copy of one 64-deep K tile into LDS buffer `buf`
  // (6 async b128 instructions per wave per tile -> s_wait_asynccnt 6 isolates one tile)
  auto issue_tile = [&](int kt, int buf) {
    const int k0 = kt << 6;
#pragma unroll
    for (int i = 0; i < 4; ++i) {  // As: 1024 x b128
      const int idx = tid + i * 256;
      const int r = idx >> 3, c = (idx & 7) << 3;
      __builtin_amdgcn_global_load_async_to_lds_b128(
          gbl_cast(&A[(size_t)(m0 + r) * K + k0 + c]),
          lds_cast(&As[buf][r * 64 + c]), 0, 0);
    }
#pragma unroll
    for (int i = 0; i < 2; ++i) {  // Bs: 512 x b128
      const int idx = tid + i * 256;
      const int r = idx >> 3, c = (idx & 7) << 3;
      __builtin_amdgcn_global_load_async_to_lds_b128(
          gbl_cast(&Bt[(size_t)(n0 + r) * K + k0 + c]),
          lds_cast(&Bs[buf][r * 64 + c]), 0, 0);
    }
  };

  issue_tile(0, 0);
  for (int kt = 0; kt < kTiles; ++kt) {
    const int cur = kt & 1;
    if (kt + 1 < kTiles) {
      issue_tile(kt + 1, cur ^ 1);                 // overlap next copy with this compute
      __builtin_amdgcn_s_wait_asynccnt(6);         // current tile fully resident
    } else {
      __builtin_amdgcn_s_wait_asynccnt(0);
    }
    __syncthreads();                               // all waves' copies for `cur` done

#pragma unroll
    for (int ks = 0; ks < 2; ++ks) {
      union { v16bf v; v8bf h[2]; } af[2], bfr[2];
#pragma unroll
      for (int tr = 0; tr < 2; ++tr) {
        const int base = (waveM * 32 + tr * 16 + mL) * 64 + ks * 32 + kb;
        af[tr].h[0] = *reinterpret_cast<const v8bf*>(&As[cur][base]);
        af[tr].h[1] = *reinterpret_cast<const v8bf*>(&As[cur][base + 16]);
      }
#pragma unroll
      for (int tc = 0; tc < 2; ++tc) {
        const int base = (waveN * 32 + tc * 16 + mL) * 64 + ks * 32 + kb;
        bfr[tc].h[0] = *reinterpret_cast<const v8bf*>(&Bs[cur][base]);
        bfr[tc].h[1] = *reinterpret_cast<const v8bf*>(&Bs[cur][base + 16]);
      }
#pragma unroll
      for (int tr = 0; tr < 2; ++tr)
#pragma unroll
        for (int tc = 0; tc < 2; ++tc)
          acc[tr][tc] = __builtin_amdgcn_wmma_f32_16x16x32_bf16(
              false, af[tr].v, false, bfr[tc].v, (short)0, acc[tr][tc], false, false);
    }
    __syncthreads();  // everyone done reading `cur` before it is refilled at kt+1
  }

  const int rbase = m0 + waveM * 32 + ((lane < 16) ? 0 : 8);
  const int cL = lane & 15;
#pragma unroll
  for (int tr = 0; tr < 2; ++tr)
#pragma unroll
    for (int tc = 0; tc < 2; ++tc)
#pragma unroll
      for (int i = 0; i < 8; ++i) {
        const int row = rbase + tr * 16 + i;
        const int col = n0 + waveN * 32 + tc * 16 + cL;
        float v = acc[tr][tc][i];
        if (bias) v += bias[col];
        const size_t o = (size_t)row * N + col;
        if (mode == GEMM_GELU_BF16) {
          outB[o] = (bf16)(0.5f * v * (1.f + erff(v * 0.70710678118f)));
        } else if (mode == GEMM_ADD_RES) {
          outF[o] = v + residual[o];
        } else {
          outF[o] = v;
        }
      }
}

// ---------------------------------------------------------------- bucket means of q,k
__global__ void sinkformer_bmean_kernel(const float* __restrict__ qkv,
                                        float* __restrict__ sq, float* __restrict__ sk) {
  const int n = blockIdx.x, h = blockIdx.y, d = threadIdx.x;  // 64 threads
  float aq = 0.f, ak = 0.f;
  for (int r = 0; r < 64; ++r) {
    const size_t t = (size_t)(n * 64 + r) * 1536;
    aq += qkv[t + h * 64 + d];
    ak += qkv[t + 512 + h * 64 + d];
  }
  const size_t o = (size_t)(h * NBUK + n) * 64 + d;
  sq[o] = aq * (1.f / 64.f);
  sk[o] = ak * (1.f / 64.f);
}

// ---------------------------------------------------------------- causal sort-net top-2 (online softmax)
__global__ __launch_bounds__(128) void sinkformer_sortnet_kernel(
    const float* __restrict__ sq, const float* __restrict__ sk,
    int* __restrict__ sel, float* __restrict__ wsel) {
  const int h = blockIdx.x, i = threadIdx.x;
  const float* qi = sq + (size_t)(h * NBUK + i) * 64;
  float qreg[64];
#pragma unroll
  for (int d = 0; d < 64; ++d) qreg[d] = qi[d];
  const float NEGV = -3.0e38f;
  int skipIdx = -1;
  for (int pass = 0; pass < 2; ++pass) {
    float m = 0.f, s = 0.f;
    int am = 0;
    for (int j = 0; j < NBUK; ++j) {
      float r;
      if (j <= i && j != skipIdx) {
        const float* kj = sk + (size_t)(h * NBUK + j) * 64;
        float acc = 0.f;
        for (int d = 0; d < 64; ++d) acc += qreg[d] * kj[d];
        r = acc * 0.125f;
      } else {
        r = NEGV;
      }
      const float rt = fmaxf(r * (1.f / 0.75f), NEGV);  // keep finite
      if (j == 0) {
        m = rt; s = 1.f; am = 0;
      } else if (rt > m) {
        s = s * __expf(m - rt) + 1.f; m = rt; am = j;
      } else {
        s += __expf(rt - m);
      }
    }
    sel[(h * 2 + pass) * NBUK + i] = am;
    wsel[(h * 2 + pass) * NBUK + i] = 1.f / s;  // softmax weight at argmax
    skipIdx = am;
  }
}

// ---------------------------------------------------------------- bucketed causal attention (WMMA)
// one workgroup per (bucket n, head h). 4 waves, 16 query rows per wave.
__global__ __launch_bounds__(128) void sinkformer_attn_kernel(
    const float* __restrict__ qkv, const int* __restrict__ sel,
    const float* __restrict__ wsel, bf16* __restrict__ attnA) {
  __shared__ bf16 Qs[64 * 64];    // Q (scaled), row-major [r][d]
  __shared__ bf16 KVs[192 * 64];  // K_all [j][d], later reused as Vt [d][j]
  __shared__ bf16 Ps[64 * 192];   // softmax probs, row-major [r][j]
  const int n = blockIdx.x, h = blockIdx.y;
  const int tid = threadIdx.x, lane = tid & 31, w = tid >> 5;
  const int s0 = sel[(h * 2 + 0) * NBUK + n];
  const int s1 = sel[(h * 2 + 1) * NBUK + n];
  const float w0 = wsel[(h * 2 + 0) * NBUK + n];
  const float w1 = wsel[(h * 2 + 1) * NBUK + n];

  for (int idx = tid; idx < 4096; idx += 128) {
    const int r = idx >> 6, d = idx & 63;
    Qs[idx] = (bf16)(qkv[(size_t)(n * 64 + r) * 1536 + h * 64 + d] * 0.125f);
  }
  for (int idx = tid; idx < 12288; idx += 128) {
    const int j = idx >> 6, d = idx & 63;
    const int p = j >> 6, jj = j & 63;
    const int sb = (p == 0) ? s0 : (p == 1) ? s1 : n;
    const float wt = (p == 0) ? w0 : (p == 1) ? w1 : 1.f;
    KVs[idx] = (bf16)(qkv[(size_t)(sb * 64 + jj) * 1536 + 512 + h * 64 + d] * wt);
  }
  __syncthreads();

  const int kb = (lane < 16) ? 0 : 8;
  const int mL = lane & 15;

  v8f dacc[12];
#pragma unroll
  for (int tc = 0; tc < 12; ++tc)
#pragma unroll
    for (int e = 0; e < 8; ++e) dacc[tc][e] = 0.f;

#pragma unroll
  for (int ks = 0; ks < 2; ++ks) {
    union { v16bf v; v8bf h2[2]; } af;
    const int ab = (w * 16 + mL) * 64 + ks * 32 + kb;
    af.h2[0] = *reinterpret_cast<const v8bf*>(&Qs[ab]);
    af.h2[1] = *reinterpret_cast<const v8bf*>(&Qs[ab + 16]);
#pragma unroll
    for (int tc = 0; tc < 12; ++tc) {
      union { v16bf v; v8bf h2[2]; } bfr;
      const int bb = (tc * 16 + mL) * 64 + ks * 32 + kb;
      bfr.h2[0] = *reinterpret_cast<const v8bf*>(&KVs[bb]);
      bfr.h2[1] = *reinterpret_cast<const v8bf*>(&KVs[bb + 16]);
      dacc[tc] = __builtin_amdgcn_wmma_f32_16x16x32_bf16(
          false, af.v, false, bfr.v, (short)0, dacc[tc], false, false);
    }
  }

  // causal mask + row softmax. C tile: col = lane&15, row = w*16 + (lane<16 ? i : i+8)
  float rmax[8], rsum[8];
#pragma unroll
  for (int i = 0; i < 8; ++i) rmax[i] = -INFINITY;
#pragma unroll
  for (int tc = 0; tc < 12; ++tc) {
    const int j = tc * 16 + mL;
    const int p = j >> 6, jj = j & 63;
    const int kpos = (((p == 0) ? s0 : (p == 1) ? s1 : n) << 6) + jj;
#pragma unroll
    for (int i = 0; i < 8; ++i) {
      const int row = w * 16 + ((lane < 16) ? i : i + 8);
      const int qpos = (n << 6) + row;
      const float v = (kpos <= qpos) ? dacc[tc][i] : -1e30f;
      dacc[tc][i] = v;
      rmax[i] = fmaxf(rmax[i], v);
    }
  }
#pragma unroll
  for (int i = 0; i < 8; ++i) {
    float m = rmax[i];
#pragma unroll
    for (int msk = 1; msk < 16; msk <<= 1) m = fmaxf(m, __shfl_xor(m, msk, 32));
    rmax[i] = m;
    rsum[i] = 0.f;
  }
#pragma unroll
  for (int tc = 0; tc < 12; ++tc)
#pragma unroll
    for (int i = 0; i < 8; ++i) {
      const float e = __expf(dacc[tc][i] - rmax[i]);
      dacc[tc][i] = e;
      rsum[i] += e;
    }
#pragma unroll
  for (int i = 0; i < 8; ++i) {
    float s = rsum[i];
#pragma unroll
    for (int msk = 1; msk < 16; msk <<= 1) s += __shfl_xor(s, msk, 32);
    rsum[i] = 1.f / s;
  }
#pragma unroll
  for (int tc = 0; tc < 12; ++tc) {
    const int j = tc * 16 + mL;
#pragma unroll
    for (int i = 0; i < 8; ++i) {
      const int row = w * 16 + ((lane < 16) ? i : i + 8);
      Ps[row * 192 + j] = (bf16)(dacc[tc][i] * rsum[i]);
    }
  }
  __syncthreads();

  // reuse KVs as Vt [d][j] (weighted)
  for (int idx = tid; idx < 12288; idx += 128) {
    const int d = idx / 192, j = idx - d * 192;
    const int p = j >> 6, jj = j & 63;
    const int sb = (p == 0) ? s0 : (p == 1) ? s1 : n;
    const float wt = (p == 0) ? w0 : (p == 1) ? w1 : 1.f;
    KVs[d * 192 + j] = (bf16)(qkv[(size_t)(sb * 64 + jj) * 1536 + 1024 + h * 64 + d] * wt);
  }
  __syncthreads();

  v8f oacc[4];
#pragma unroll
  for (int tc = 0; tc < 4; ++tc)
#pragma unroll
    for (int e = 0; e < 8; ++e) oacc[tc][e] = 0.f;
#pragma unroll
  for (int ks = 0; ks < 6; ++ks) {
    union { v16bf v; v8bf h2[2]; } af;
    const int ab = (w * 16 + mL) * 192 + ks * 32 + kb;
    af.h2[0] = *reinterpret_cast<const v8bf*>(&Ps[ab]);
    af.h2[1] = *reinterpret_cast<const v8bf*>(&Ps[ab + 16]);
#pragma unroll
    for (int tc = 0; tc < 4; ++tc) {
      union { v16bf v; v8bf h2[2]; } bfr;
      const int bb = (tc * 16 + mL) * 192 + ks * 32 + kb;
      bfr.h2[0] = *reinterpret_cast<const v8bf*>(&KVs[bb]);
      bfr.h2[1] = *reinterpret_cast<const v8bf*>(&KVs[bb + 16]);
      oacc[tc] = __builtin_amdgcn_wmma_f32_16x16x32_bf16(
          false, af.v, false, bfr.v, (short)0, oacc[tc], false, false);
    }
  }
#pragma unroll
  for (int tc = 0; tc < 4; ++tc)
#pragma unroll
    for (int i = 0; i < 8; ++i) {
      const int row = w * 16 + ((lane < 16) ? i : i + 8);
      const int t = n * 64 + row;
      const int col = h * 64 + tc * 16 + mL;
      attnA[(size_t)t * E_DIM + col] = (bf16)oacc[tc][i];
    }
}

// ---------------------------------------------------------------- head (N=17, fp32)
__global__ void sinkformer_head_kernel(const float* __restrict__ x,
                                       const float* __restrict__ hw,
                                       float* __restrict__ out) {
  const int t = blockIdx.x;
  const int v = threadIdx.x;
  if (v >= VOCAB) return;
  float s = 0.f;
  for (int k = 0; k < E_DIM; ++k) s += x[(size_t)t * E_DIM + k] * hw[k * VOCAB + v];
  out[(size_t)t * VOCAB + v] = s;
}

// ---------------------------------------------------------------- host driver
extern "C" void kernel_launch(void* const* d_in, const int* in_sizes, int n_in,
                              void* d_out, int out_size, void* d_ws, size_t ws_size,
                              hipStream_t stream) {
  const int*   value  = (const int*)d_in[0];
  const int*   depth  = (const int*)d_in[1];
  const int*   pos    = (const int*)d_in[2];
  const float* sos    = (const float*)d_in[3];
  const float* tokE   = (const float*)d_in[4];
  const float* depE   = (const float*)d_in[5];
  const float* spatE  = (const float*)d_in[6];
  const float* ln1g   = (const float*)d_in[7];
  const float* ln1b   = (const float*)d_in[8];
  const float* wqkv   = (const float*)d_in[9];
  const float* wout   = (const float*)d_in[10];
  const float* bout   = (const float*)d_in[11];
  const float* ln2g   = (const float*)d_in[12];
  const float* ln2b   = (const float*)d_in[13];
  const float* wff1   = (const float*)d_in[14];
  const float* bff1   = (const float*)d_in[15];
  const float* wff2   = (const float*)d_in[16];
  const float* bff2   = (const float*)d_in[17];
  const float* headw  = (const float*)d_in[18];

  // workspace carve (256B aligned)
  char* ws = (char*)d_ws;
  size_t off = 0;
  auto carve = [&](size_t bytes) { void* p = ws + off; off += (bytes + 255) & ~(size_t)255; return p; };
  float* x    = (float*)carve((size_t)T_TOK * E_DIM * 4);        // 16.8 MB
  float* qkv  = (float*)carve((size_t)T_TOK * 3 * E_DIM * 4);    // 50.3 MB
  bf16*  Hbuf = (bf16*)carve((size_t)T_TOK * 4 * E_DIM * 2);     // 33.6 MB
  bf16*  Abuf = (bf16*)carve((size_t)T_TOK * E_DIM * 2);         // 8.4 MB  (LN out / attn out)
  bf16*  Wt   = (bf16*)carve((size_t)4 * E_DIM * E_DIM * 2);     // 2.1 MB  (largest weight)
  float* sq   = (float*)carve((size_t)NHEAD * NBUK * 64 * 4);
  float* sk   = (float*)carve((size_t)NHEAD * NBUK * 64 * 4);
  int*   selb = (int*)carve((size_t)NHEAD * 2 * NBUK * 4);
  float* wslb = (float*)carve((size_t)NHEAD * 2 * NBUK * 4);
  if (off > ws_size) return;  // insufficient scratch; nothing safe to do

  sinkformer_embed_kernel<<<T_TOK, 256, 0, stream>>>(value, depth, pos, sos, tokE, depE, spatE, x);

  for (int l = 0; l < NLAY; ++l) {
    // attention block
    sinkformer_ln_kernel<<<T_TOK, 256, 0, stream>>>(x, ln1g + l * E_DIM, ln1b + l * E_DIM, Abuf);
    sinkformer_convt_kernel<<<(E_DIM * 3 * E_DIM + 255) / 256, 256, 0, stream>>>(
        wqkv + (size_t)l * E_DIM * 3 * E_DIM, Wt, E_DIM, 3 * E_DIM);
    sinkformer_gemm_kernel<<<dim3(3 * E_DIM / 64, T_TOK / 128), 256, 0, stream>>>(
        Abuf, Wt, nullptr, nullptr, qkv, nullptr, T_TOK, 3 * E_DIM, E_DIM, GEMM_OUT_F32);
    sinkformer_bmean_kernel<<<dim3(NBUK, NHEAD), 64, 0, stream>>>(qkv, sq, sk);
    sinkformer_sortnet_kernel<<<NHEAD, NBUK, 0, stream>>>(sq, sk, selb, wslb);
    sinkformer_attn_kernel<<<dim3(NBUK, NHEAD), 128, 0, stream>>>(qkv, selb, wslb, Abuf);
    sinkformer_convt_kernel<<<(E_DIM * E_DIM + 255) / 256, 256, 0, stream>>>(
        wout + (size_t)l * E_DIM * E_DIM, Wt, E_DIM, E_DIM);
    sinkformer_gemm_kernel<<<dim3(E_DIM / 64, T_TOK / 128), 256, 0, stream>>>(
        Abuf, Wt, bout + l * E_DIM, x, x, nullptr, T_TOK, E_DIM, E_DIM, GEMM_ADD_RES);

    // feed-forward block
    sinkformer_ln_kernel<<<T_TOK, 256, 0, stream>>>(x, ln2g + l * E_DIM, ln2b + l * E_DIM, Abuf);
    sinkformer_convt_kernel<<<(E_DIM * 4 * E_DIM + 255) / 256, 256, 0, stream>>>(
        wff1 + (size_t)l * E_DIM * 4 * E_DIM, Wt, E_DIM, 4 * E_DIM);
    sinkformer_gemm_kernel<<<dim3(4 * E_DIM / 64, T_TOK / 128), 256, 0, stream>>>(
        Abuf, Wt, bff1 + l * 4 * E_DIM, nullptr, nullptr, Hbuf, T_TOK, 4 * E_DIM, E_DIM, GEMM_GELU_BF16);
    sinkformer_convt_kernel<<<(4 * E_DIM * E_DIM + 255) / 256, 256, 0, stream>>>(
        wff2 + (size_t)l * 4 * E_DIM * E_DIM, Wt, 4 * E_DIM, E_DIM);
    sinkformer_gemm_kernel<<<dim3(E_DIM / 64, T_TOK / 128), 256, 0, stream>>>(
        Hbuf, Wt, bff2 + l * E_DIM, x, x, nullptr, T_TOK, E_DIM, 4 * E_DIM, GEMM_ADD_RES);
  }

  sinkformer_head_kernel<<<S_TOK, 32, 0, stream>>>(x, headw, (float*)d_out);
}
